// TransformerLayer_43628277793447
// MI455X (gfx1250) — matrix operand order repaired and checked by reading, not compiled, not run on hardware
//
#include <hip/hip_runtime.h>
#include <hip/hip_bf16.h>

typedef __bf16 bf16_t;
typedef __attribute__((ext_vector_type(16))) __bf16 v16bf;
typedef __attribute__((ext_vector_type(8)))  __bf16 v8bf;
typedef __attribute__((ext_vector_type(8)))  float  v8f;
typedef __attribute__((ext_vector_type(4)))  int    v4i;

#define TB  4
#define TS  2048
#define TE  1024
#define TFF 4096
#define TBS (TB*TS)

__device__ __forceinline__ bf16_t f2bf(float f) { return (bf16_t)f; }

// ---------------------------------------------------------------------------
// Async global -> LDS copy of one 16-byte chunk (CDNA5 ASYNCcnt path).
// Builtin expects (v4i AS1*, v4i AS3*, imm offset, imm cpol).
// Falls back to a synchronous copy if the builtin is unavailable.
// ---------------------------------------------------------------------------
#if __has_builtin(__builtin_amdgcn_global_load_async_to_lds_b128)
#define HAS_ASYNC_LDS 1
#else
#define HAS_ASYNC_LDS 0
#endif

__device__ __forceinline__ void cp16_async(const bf16_t* __restrict__ g,
                                           bf16_t* l) {
#if HAS_ASYNC_LDS
    __builtin_amdgcn_global_load_async_to_lds_b128(
        (__attribute__((address_space(1))) v4i*)g,
        (__attribute__((address_space(3))) v4i*)l, 0, 0);
#else
    *(v8bf*)l = *(const v8bf*)g;
#endif
}

__device__ __forceinline__ void wait_async0() {
#if HAS_ASYNC_LDS
#if __has_builtin(__builtin_amdgcn_s_wait_asynccnt)
    __builtin_amdgcn_s_wait_asynccnt(0);
#else
    asm volatile("s_wait_asynccnt 0x0" ::: "memory");
#endif
#endif
}

// ---------------------------------------------------------------------------
// Weight convert + transpose: dst[n*K + k] = bf16(src[k*N + n])
// ---------------------------------------------------------------------------
__global__ void wtrans_kernel(const float* __restrict__ src,
                              bf16_t* __restrict__ dst, int K, int N) {
    long idx = (long)blockIdx.x * blockDim.x + threadIdx.x;
    int n = (int)(idx / K);
    int k = (int)(idx % K);
    dst[(long)n * K + k] = f2bf(src[(long)k * N + n]);
}

// ---------------------------------------------------------------------------
// Extract V from packed qkv (bf16) and transpose: vT[b][e][s]
// ---------------------------------------------------------------------------
__global__ void vtrans_kernel(const bf16_t* __restrict__ qkv,
                              bf16_t* __restrict__ vT) {
    long idx = (long)blockIdx.x * blockDim.x + threadIdx.x;
    int b   = (int)(idx / ((long)TE * TS));
    int rem = (int)(idx % ((long)TE * TS));
    int e = rem / TS;
    int k = rem % TS;
    vT[(long)b * TE * TS + (long)e * TS + k] =
        qkv[((long)b * TS + k) * (3 * TE) + 2 * TE + e];
}

// ---------------------------------------------------------------------------
// LayerNorm: one 256-thread block per row of E=1024; fp32 in -> bf16 out
// ---------------------------------------------------------------------------
__global__ void ln_kernel(const float* __restrict__ in,
                          const float* __restrict__ g,
                          const float* __restrict__ be,
                          bf16_t* __restrict__ out) {
    __shared__ float red[256];
    int row = blockIdx.x, t = threadIdx.x;
    const float* p = in + (long)row * TE;
    float v[4], s = 0.f;
#pragma unroll
    for (int i = 0; i < 4; i++) { v[i] = p[t + i * 256]; s += v[i]; }
    red[t] = s; __syncthreads();
    for (int o = 128; o >= 1; o >>= 1) { if (t < o) red[t] += red[t + o]; __syncthreads(); }
    float mu = red[0] * (1.0f / TE);
    __syncthreads();
    float q = 0.f;
#pragma unroll
    for (int i = 0; i < 4; i++) { float d = v[i] - mu; q += d * d; }
    red[t] = q; __syncthreads();
    for (int o = 128; o >= 1; o >>= 1) { if (t < o) red[t] += red[t + o]; __syncthreads(); }
    float rstd = rsqrtf(red[0] * (1.0f / TE) + 1e-5f);
#pragma unroll
    for (int i = 0; i < 4; i++) {
        int c = t + i * 256;
        out[(long)row * TE + c] = f2bf((v[i] - mu) * rstd * g[c] + be[c]);
    }
}

// ---------------------------------------------------------------------------
// Causal scaled softmax over rows of S=2048; fp32 scores -> bf16 probs
// ---------------------------------------------------------------------------
__global__ void softmax_kernel(const float* __restrict__ scores,
                               bf16_t* __restrict__ probs) {
    __shared__ float red[256];
    int rowg = blockIdx.x;
    int q = rowg % TS;
    int t = threadIdx.x;
    const float* p = scores + (long)rowg * TS;
    const float scale = 0.03125f;  // 1/sqrt(1024)
    float v[8], mx = -__builtin_inff();
#pragma unroll
    for (int i = 0; i < 8; i++) {
        int c = t + i * 256;
        v[i] = (c <= q) ? p[c] * scale : -__builtin_inff();
        mx = fmaxf(mx, v[i]);
    }
    red[t] = mx; __syncthreads();
    for (int o = 128; o >= 1; o >>= 1) { if (t < o) red[t] = fmaxf(red[t], red[t + o]); __syncthreads(); }
    mx = red[0]; __syncthreads();
    float s = 0.f;
#pragma unroll
    for (int i = 0; i < 8; i++) {
        int c = t + i * 256;
        v[i] = (c <= q) ? __expf(v[i] - mx) : 0.f;
        s += v[i];
    }
    red[t] = s; __syncthreads();
    for (int o = 128; o >= 1; o >>= 1) { if (t < o) red[t] += red[t + o]; __syncthreads(); }
    float inv = 1.0f / red[0];
#pragma unroll
    for (int i = 0; i < 8; i++) {
        int c = t + i * 256;
        probs[(long)rowg * TS + c] = f2bf(v[i] * inv);
    }
}

// ---------------------------------------------------------------------------
// Generic bf16 WMMA GEMM: C[b] = A[b] (MxK, row-major) * Bt[b]^T (Bt is NxK)
// Block tile 128x256, 8 waves of 64x64, K-step 32.
// Double-buffered LDS, staged with GLOBAL_LOAD_ASYNC_TO_LDS_B128 (ASYNCcnt).
// mode 0: bf16 store (+bias)     mode 1: bf16 store, bias+ReLU
// mode 2: fp32 store, bias(+opt) + residual    mode 3: raw fp32 store
// causal_skip:  drop blocks fully above the diagonal (QK^T)
// causal_klimit: truncate K loop at m0+BM (PV: probs are 0 beyond diagonal)
// ---------------------------------------------------------------------------
__global__ __launch_bounds__(256)
void gemm_bf16_kernel(const bf16_t* __restrict__ A, long sAb, int lda,
                      const bf16_t* __restrict__ Bt, long sBb, int ldb,
                      const float* __restrict__ bias,
                      const float* __restrict__ resid, long sRb, int ldr,
                      float* __restrict__ outf, bf16_t* __restrict__ outb,
                      long sOb, int ldo,
                      int K, int mode, int causal_skip, int causal_klimit) {
    constexpr int BM = 128, BN = 256, BK = 32;
    __shared__ bf16_t As[2][BM][BK];
    __shared__ bf16_t Bs[2][BN][BK];

    int m0 = blockIdx.y * BM;
    int n0 = blockIdx.x * BN;
    int b  = blockIdx.z;
    if (causal_skip && n0 > m0 + BM - 1) return;

    A  += (long)b * sAb;
    Bt += (long)b * sBb;

    int t = threadIdx.x;
    int lane = t & 31;
    int w = t >> 5;
    int wm = w >> 2;   // 0..1
    int wn = w & 3;    // 0..3
    int l15 = lane & 15;
    int lh  = lane >> 4;

    v8f acc[4][4];
#pragma unroll
    for (int i = 0; i < 4; i++)
#pragma unroll
        for (int j = 0; j < 4; j++)
            acc[i][j] = (v8f){0.f, 0.f, 0.f, 0.f, 0.f, 0.f, 0.f, 0.f};

    int ar = t >> 1;            // A rows 0..127, 32B per thread
    int ak = (t & 1) * 16;
    int br = t;                 // B rows 0..255, 64B per thread

    const bf16_t* gA = A + (long)(m0 + ar) * lda + ak;
    const bf16_t* gB = Bt + (long)(n0 + br) * ldb;

    int kEnd = causal_klimit ? ((m0 + BM < K) ? (m0 + BM) : K) : K;

    auto stage = [&](int buf, int k0) {
        cp16_async(gA + k0,      &As[buf][ar][ak]);
        cp16_async(gA + k0 + 8,  &As[buf][ar][ak + 8]);
        cp16_async(gB + k0,      &Bs[buf][br][0]);
        cp16_async(gB + k0 + 8,  &Bs[buf][br][8]);
        cp16_async(gB + k0 + 16, &Bs[buf][br][16]);
        cp16_async(gB + k0 + 24, &Bs[buf][br][24]);
    };

    stage(0, 0);
    int cur = 0;

    for (int k0 = 0; k0 < kEnd; k0 += BK) {
        wait_async0();          // staged tile for this iteration is in LDS
        __syncthreads();        // (compiler also drains dscnt: prev frag reads done)
        if (k0 + BK < kEnd) stage(cur ^ 1, k0 + BK);

        v16bf afr[4], bfr[4];
#pragma unroll
        for (int i = 0; i < 4; i++) {
            // 16-bit A layout: lanes 0-15 K0..7/K16..23, lanes 16-31 K8..15/K24..31
            const bf16_t* pa = &As[cur][wm * 64 + i * 16 + l15][lh * 8];
            v8bf lo = *(const v8bf*)pa;
            v8bf hi = *(const v8bf*)(pa + 16);
            afr[i] = __builtin_shufflevector(lo, hi, 0, 1, 2, 3, 4, 5, 6, 7,
                                             8, 9, 10, 11, 12, 13, 14, 15);
        }
#pragma unroll
        for (int j = 0; j < 4; j++) {
            // B layout: lanes 0-15 K0..7/K16..23, lanes 16-31 K8..15/K24..31
            const bf16_t* pb = &Bs[cur][wn * 64 + j * 16 + l15][lh * 8];
            bfr[j] = __builtin_shufflevector(*(const v8bf*)pb, *(const v8bf*)(pb + 16),
                                             0, 1, 2, 3, 4, 5, 6, 7, 8, 9, 10, 11, 12, 13, 14, 15);
        }
#pragma unroll
        for (int i = 0; i < 4; i++)
#pragma unroll
            for (int j = 0; j < 4; j++)
                acc[i][j] = __builtin_amdgcn_wmma_f32_16x16x32_bf16(
                    false, afr[i], false, bfr[j], (short)0, acc[i][j], false, false);
        cur ^= 1;
    }

    long ob = (long)b * sOb;
#pragma unroll
    for (int i = 0; i < 4; i++) {
#pragma unroll
        for (int j = 0; j < 4; j++) {
            int mt = m0 + wm * 64 + i * 16 + lh * 8;
            int nn = n0 + wn * 64 + j * 16 + l15;
            float bs = bias ? bias[nn] : 0.f;
#pragma unroll
            for (int r = 0; r < 8; r++) {
                int m = mt + r;
                float vv = acc[i][j][r];
                long off = ob + (long)m * ldo + nn;
                if (mode == 0) {
                    outb[off] = f2bf(vv + bs);
                } else if (mode == 1) {
                    outb[off] = f2bf(fmaxf(vv + bs, 0.f));
                } else if (mode == 2) {
                    outf[off] = vv + bs + resid[(long)b * sRb + (long)m * ldr + nn];
                } else {
                    outf[off] = vv;
                }
            }
        }
    }
}

// ---------------------------------------------------------------------------
extern "C" void kernel_launch(void* const* d_in, const int* in_sizes, int n_in,
                              void* d_out, int out_size, void* d_ws, size_t ws_size,
                              hipStream_t stream) {
    (void)in_sizes; (void)n_in; (void)out_size; (void)ws_size;
    const float* x     = (const float*)d_in[0];
    const float* w_qkv = (const float*)d_in[1];
    const float* b_qkv = (const float*)d_in[2];
    const float* w_a   = (const float*)d_in[3];
    const float* b_a   = (const float*)d_in[4];
    const float* w_b   = (const float*)d_in[5];
    const float* b_b   = (const float*)d_in[6];
    const float* g1    = (const float*)d_in[7];
    const float* be1   = (const float*)d_in[8];
    const float* g2    = (const float*)d_in[9];
    const float* be2   = (const float*)d_in[10];
    float* out = (float*)d_out;

    char* ws = (char*)d_ws;
    size_t o = 0;
    bf16_t* wqkvT = (bf16_t*)(ws + o); o += (size_t)3 * TE * TE * 2;    // [3E][E]
    bf16_t* waT   = (bf16_t*)(ws + o); o += (size_t)TFF * TE * 2;       // [FF][E]
    bf16_t* wbT   = (bf16_t*)(ws + o); o += (size_t)TE * TFF * 2;       // [E][FF]
    bf16_t* h12   = (bf16_t*)(ws + o); o += (size_t)TBS * TE * 2;       // h1 then h2
    char*   qkvR  = ws + o;            o += (size_t)TBS * 3 * TE * 2;   // qkv bf16; later x2 fp32
    bf16_t* qkv   = (bf16_t*)qkvR;
    float*  x2    = (float*)qkvR;      // reuse: qkv dead once probs/vT exist
    char*   scR   = ws + o;            o += (size_t)TB * TS * TS * 4;   // scores f32; later act bf16
    float*  scores = (float*)scR;
    bf16_t* act    = (bf16_t*)scR;     // reuse: scores dead after softmax (same 64MB)
    bf16_t* probs  = (bf16_t*)(ws + o); o += (size_t)TB * TS * TS * 2;
    bf16_t* vT     = (bf16_t*)(ws + o); o += (size_t)TB * TE * TS * 2;

    dim3 blk(256);

    // 1. weight convert+transpose to bf16 N-major
    wtrans_kernel<<<(TE * 3 * TE) / 256, blk, 0, stream>>>(w_qkv, wqkvT, TE, 3 * TE);
    wtrans_kernel<<<(TE * TFF) / 256,    blk, 0, stream>>>(w_a,   waT,   TE, TFF);
    wtrans_kernel<<<(TFF * TE) / 256,    blk, 0, stream>>>(w_b,   wbT,   TFF, TE);

    // 2. LN1(x) -> h1 (bf16)
    ln_kernel<<<TBS, blk, 0, stream>>>(x, g1, be1, h12);

    // 3. qkv = h1 @ w_qkv + b_qkv  (bf16 out)
    gemm_bf16_kernel<<<dim3(3 * TE / 256, TBS / 128, 1), blk, 0, stream>>>(
        h12, 0, TE, wqkvT, 0, TE, b_qkv,
        nullptr, 0, 0, nullptr, qkv, 0, 3 * TE, TE, /*mode*/0, 0, 0);

    // 4. vT[b][e][s]
    vtrans_kernel<<<((long)TB * TE * TS) / 256, blk, 0, stream>>>(qkv, vT);

    // 5. scores = q @ k^T  (causal tile skip, raw fp32)
    gemm_bf16_kernel<<<dim3(TS / 256, TS / 128, TB), blk, 0, stream>>>(
        qkv, (long)TS * 3 * TE, 3 * TE, qkv + TE, (long)TS * 3 * TE, 3 * TE,
        nullptr, nullptr, 0, 0, scores, nullptr, (long)TS * TS, TS,
        TE, /*mode*/3, /*causal_skip*/1, 0);

    // 6. probs = softmax(mask(scores)/32)  (bf16)
    softmax_kernel<<<TB * TS, blk, 0, stream>>>(scores, probs);

    // 7. x2 = x + probs @ v   (fp32, residual add; K truncated at diagonal)
    gemm_bf16_kernel<<<dim3(TE / 256, TS / 128, TB), blk, 0, stream>>>(
        probs, (long)TS * TS, TS, vT, (long)TE * TS, TS,
        nullptr, x, (long)TS * TE, TE, x2, nullptr, (long)TS * TE, TE,
        TS, /*mode*/2, 0, /*causal_klimit*/1);

    // 8. LN2(x2) -> h2 (bf16, reuses h1 buffer)
    ln_kernel<<<TBS, blk, 0, stream>>>(x2, g2, be2, h12);

    // 9. act = relu(h2 @ w_a + b_a)  (bf16)
    gemm_bf16_kernel<<<dim3(TFF / 256, TBS / 128, 1), blk, 0, stream>>>(
        h12, 0, TE, waT, 0, TE, b_a,
        nullptr, 0, 0, nullptr, act, 0, TFF, TE, /*mode*/1, 0, 0);

    // 10. out = x2 + act @ w_b + b_b  (fp32)
    gemm_bf16_kernel<<<dim3(TE / 256, TBS / 128, 1), blk, 0, stream>>>(
        act, 0, TFF, wbT, 0, TFF, b_b,
        x2, 0, TE, out, nullptr, 0, TE, TFF, /*mode*/2, 0, 0);
}
